// HierarchicalGRU_70712341561351
// MI455X (gfx1250) — matrix operand order repaired and checked by reading, not compiled, not run on hardware
//
#include <hip/hip_runtime.h>

// Problem dims
#define BDIM  16384
#define INDIM 4096
#define HDIM  128
#define NOPTD 64

// Output layout (flat f32, concatenated in return order)
#define OFF_V0  0
#define OFF_TP  (BDIM * 64)
#define OFF_V1  (BDIM * 65)
#define OFF_H0N (BDIM * 129)
#define OFF_H1N (BDIM * 257)

typedef __attribute__((ext_vector_type(16))) _Float16 v16h;
typedef __attribute__((ext_vector_type(8)))  _Float16 v8h;
typedef __attribute__((ext_vector_type(4)))  _Float16 v4h;
typedef __attribute__((ext_vector_type(2)))  _Float16 v2h;
typedef __attribute__((ext_vector_type(8)))  float    v8f;

__device__ __forceinline__ v8f zero8() {
    v8f z;
#pragma unroll
    for (int i = 0; i < 8; ++i) z[i] = 0.0f;
    return z;
}

__device__ __forceinline__ v16h cat8(v8h a, v8h b) {
    v16h r;
#pragma unroll
    for (int i = 0; i < 8; ++i) { r[i] = a[i]; r[i + 8] = b[i]; }
    return r;
}

// A-fragment (16 rows x 32 K, f16 row-major source, stride in halves).
// Lane layout per ISA 7.12.2: lanes 0-15 row=lane, K in {hi*8..hi*8+7, 16+hi*8..+7}.
__device__ __forceinline__ v16h afrag(const _Float16* base, int strideH, int lane16, int hi) {
    const _Float16* rp = base + lane16 * strideH + hi * 8;
    return cat8(*(const v8h*)rp, *(const v8h*)(rp + 16));
}

// B-fragment (32 K x 16 N) from transposed (N-major) f16 image: row=N, K in {hi*8..hi*8+7, 16+hi*8..+7}.
__device__ __forceinline__ v16h bfrag(const _Float16* baseNT, int strideH, int lane16, int hi) {
    const _Float16* rp = baseNT + lane16 * strideH + hi * 8;
    return cat8(*(const v8h*)rp, *(const v8h*)(rp + 16));
}

// A-fragment built from an f32 row-major source (on-the-fly f32->f16).
__device__ __forceinline__ v16h afrag_f32(const float* rowbase, int hi, int k0) {
    v16h r;
#pragma unroll
    for (int i = 0; i < 8; ++i) {
        int k = k0 + 2 * i + ((i >= 4) ? 8 : 0) + hi * 8;
        float2 d = *(const float2*)(rowbase + k);
        r[2 * i]     = (_Float16)d.x;
        r[2 * i + 1] = (_Float16)d.y;
    }
    return r;
}

__device__ __forceinline__ v8f wmma16(v16h a, v16h b, v8f c) {
    return __builtin_amdgcn_wmma_f32_16x16x32_f16(false, a, false, b, (short)0, c, false, false);
}

__device__ __forceinline__ float sigm(float x) { return 1.0f / (1.0f + __expf(-x)); }
__device__ __forceinline__ float tanh_fast(float x) { return 1.0f - 2.0f / (__expf(2.0f * x) + 1.0f); }

// ---------------------------------------------------------------------------
// Kernel 1: attn = tanh(obs @ Wa + ba), stored f16 to workspace.
// WG = 256 thr (8 waves); WG tile = 256 rows x 128 cols (32 rows/wave, two
// 16-row strips so each B fragment feeds two WMMAs). K tiled by 64 with
// double-buffered LDS staging: stage(k+1) is issued before compute(k) so the
// HBM reads of obs hide under the WMMAs.
// ---------------------------------------------------------------------------
__global__ __launch_bounds__(256) void attn_kernel(const float* __restrict__ obs,
                                                   const float* __restrict__ Wa,
                                                   const float* __restrict__ ba,
                                                   _Float16* __restrict__ attn_h) {
    extern __shared__ _Float16 sm1[];
    _Float16* At = sm1;                 // 2 x [256 rows][72]  (64 K + 8 pad)
    _Float16* Bt = sm1 + 2 * 256 * 72;  // 2 x [128 N][72]     transposed Wa chunk

    const int tid = threadIdx.x;
    const int wave = tid >> 5, lane = tid & 31;
    const int lane16 = lane & 15, hi = lane >> 4;
    const int row0 = blockIdx.x * 256;

    auto stage = [&](int kc, int buf) {
        _Float16* Ab = At + buf * 256 * 72;
        _Float16* Bb = Bt + buf * 128 * 72;
        // A: 256 rows x 64 K, f32->f16, 16 float4 groups per thread
#pragma unroll 4
        for (int i = 0; i < 16; ++i) {
            int g = i * 256 + tid;          // 4096 groups
            int r = g >> 4;
            int k4 = (g & 15) << 2;
            float4 d = *(const float4*)(obs + (size_t)(row0 + r) * INDIM + kc * 64 + k4);
            v4h h;
            h[0] = (_Float16)d.x; h[1] = (_Float16)d.y;
            h[2] = (_Float16)d.z; h[3] = (_Float16)d.w;
            *(v4h*)(Ab + r * 72 + k4) = h;
        }
        // B transposed: Wa[kc*64+k][n] -> Bb[n][k], two k per thread (paired store)
#pragma unroll 4
        for (int i = 0; i < 16; ++i) {
            int e = i * 256 + tid;          // 4096 k-pair x n slots
            int kp = e >> 7, n = e & 127;
            int k = kp * 2;
            float w0 = Wa[(size_t)(kc * 64 + k) * HDIM + n];
            float w1 = Wa[(size_t)(kc * 64 + k + 1) * HDIM + n];
            v2h h2; h2[0] = (_Float16)w0; h2[1] = (_Float16)w1;
            *(v2h*)(Bb + n * 72 + k) = h2;
        }
    };

    v8f acc[2][8];
#pragma unroll
    for (int s = 0; s < 2; ++s)
#pragma unroll
        for (int t = 0; t < 8; ++t) acc[s][t] = zero8();

    stage(0, 0);
    for (int kc = 0; kc < INDIM / 64; ++kc) {
        __syncthreads();                 // buf[kc&1] ready; buf[(kc+1)&1] free
        const int cur = kc & 1;
        if (kc + 1 < INDIM / 64) stage(kc + 1, cur ^ 1);

        const _Float16* Ab = At + cur * 256 * 72 + (wave * 32) * 72;
        v16h a00 = afrag(Ab, 72, lane16, hi);            // strip 0, K 0..31
        v16h a01 = afrag(Ab + 32, 72, lane16, hi);       // strip 0, K 32..63
        v16h a10 = afrag(Ab + 16 * 72, 72, lane16, hi);  // strip 1
        v16h a11 = afrag(Ab + 16 * 72 + 32, 72, lane16, hi);
        const _Float16* Bb = Bt + cur * 128 * 72;
#pragma unroll
        for (int t = 0; t < 8; ++t) {
            v16h b0 = bfrag(Bb + (t * 16) * 72, 72, lane16, hi);
            v16h b1 = bfrag(Bb + (t * 16) * 72 + 32, 72, lane16, hi);
            acc[0][t] = wmma16(a00, b0, acc[0][t]);
            acc[1][t] = wmma16(a10, b0, acc[1][t]);
            acc[0][t] = wmma16(a01, b1, acc[0][t]);
            acc[1][t] = wmma16(a11, b1, acc[1][t]);
        }
    }

    // epilogue: bias + tanh, store f16
#pragma unroll
    for (int s = 0; s < 2; ++s)
#pragma unroll
        for (int t = 0; t < 8; ++t) {
            int col = t * 16 + lane16;
            float bav = ba[col];
#pragma unroll
            for (int p = 0; p < 8; ++p) {
                int row = row0 + wave * 32 + s * 16 + p + hi * 8;
                attn_h[row * HDIM + col] = (_Float16)tanh_fast(acc[s][t][p] + bav);
            }
        }
}

// ---------------------------------------------------------------------------
// Kernel 2: GRU layer 0 (gi = attn@Wih0, gh = h0@Whh0) fused with gate math,
// gain-contrast, v0 / term heads and the term_p gather.
// All weights staged transposed f16 in LDS (~276 KB of the 320 KB WGP LDS).
// ---------------------------------------------------------------------------
__global__ __launch_bounds__(256) void gru0_kernel(const _Float16* __restrict__ attn_h,
                                                   const float* __restrict__ h0,
                                                   const float* __restrict__ Wih0,
                                                   const float* __restrict__ Whh0,
                                                   const float* __restrict__ bih0,
                                                   const float* __restrict__ bhh0,
                                                   const float* __restrict__ Wv0,
                                                   const float* __restrict__ bv0,
                                                   const float* __restrict__ Wt0,
                                                   const float* __restrict__ bt0,
                                                   const int* __restrict__ chosen,
                                                   _Float16* __restrict__ hc0_h,
                                                   float* __restrict__ out) {
    extern __shared__ _Float16 smem0[];
    _Float16* WihT = smem0;                       // 384 x 136
    _Float16* WhhT = WihT + 384 * 136;            // 384 x 136
    _Float16* WvT  = WhhT + 384 * 136;            // 64 x 136
    _Float16* WtT  = WvT + 64 * 136;              // 64 x 136
    float*    termS = (float*)(WtT + 64 * 136);   // [128][64]

    const int tid = threadIdx.x;
    const int wave = tid >> 5, lane = tid & 31;
    const int lane16 = lane & 15, hi = lane >> 4;
    const int row0 = blockIdx.x * 128;

    for (int e = tid; e < HDIM * 384; e += 256) {
        int k = e / 384, n = e % 384;
        WihT[n * 136 + k] = (_Float16)Wih0[e];
        WhhT[n * 136 + k] = (_Float16)Whh0[e];
    }
    for (int e = tid; e < HDIM * NOPTD; e += 256) {
        int k = e >> 6, n = e & 63;
        WvT[n * 136 + k] = (_Float16)Wv0[e];
        WtT[n * 136 + k] = (_Float16)Wt0[e];
    }
    __syncthreads();

    const int arow = row0 + wave * 16 + lane16;
    v16h a_at[4], a_h0[4];
#pragma unroll
    for (int c = 0; c < 4; ++c) {
        a_at[c] = afrag(attn_h + (size_t)(row0 + wave * 16) * HDIM + c * 32, HDIM, lane16, hi);
        a_h0[c] = afrag_f32(h0 + (size_t)arow * HDIM, hi, c * 32);
    }

    for (int j = 0; j < 8; ++j) {
        v8f air = zero8(), ahr = zero8(), aiz = zero8(), ahz = zero8(), ain = zero8(), ahn = zero8();
#pragma unroll
        for (int c = 0; c < 4; ++c) {
            air = wmma16(a_at[c], bfrag(WihT + (j * 16) * 136 + c * 32, 136, lane16, hi), air);
            aiz = wmma16(a_at[c], bfrag(WihT + (128 + j * 16) * 136 + c * 32, 136, lane16, hi), aiz);
            ain = wmma16(a_at[c], bfrag(WihT + (256 + j * 16) * 136 + c * 32, 136, lane16, hi), ain);
            ahr = wmma16(a_h0[c], bfrag(WhhT + (j * 16) * 136 + c * 32, 136, lane16, hi), ahr);
            ahz = wmma16(a_h0[c], bfrag(WhhT + (128 + j * 16) * 136 + c * 32, 136, lane16, hi), ahz);
            ahn = wmma16(a_h0[c], bfrag(WhhT + (256 + j * 16) * 136 + c * 32, 136, lane16, hi), ahn);
        }
        int col = j * 16 + lane16;
        float bir = bih0[col],       bhr = bhh0[col];
        float biz = bih0[128 + col], bhz = bhh0[128 + col];
        float bin = bih0[256 + col], bhn = bhh0[256 + col];
#pragma unroll
        for (int p = 0; p < 8; ++p) {
            int row = row0 + wave * 16 + p + hi * 8;
            float r = sigm(air[p] + ahr[p] + bir + bhr);
            float z = sigm(aiz[p] + ahz[p] + biz + bhz);
            float n = tanh_fast(ain[p] + bin + r * (ahn[p] + bhn));
            float hv = h0[(size_t)row * HDIM + col];
            float hn = (1.0f - z) * n + z * hv;
            out[OFF_H0N + (size_t)row * HDIM + col] = hn;
            hc0_h[(size_t)row * HDIM + col] = (_Float16)((hn - 0.5f) * 0.5f + 0.5f);
        }
    }

    asm volatile("s_wait_storecnt 0x0" ::: "memory");

    v16h a_hc[4];
#pragma unroll
    for (int c = 0; c < 4; ++c)
        a_hc[c] = afrag(hc0_h + (size_t)(row0 + wave * 16) * HDIM + c * 32, HDIM, lane16, hi);

    for (int jn = 0; jn < 4; ++jn) {
        v8f av = zero8(), at = zero8();
#pragma unroll
        for (int c = 0; c < 4; ++c) {
            av = wmma16(a_hc[c], bfrag(WvT + (jn * 16) * 136 + c * 32, 136, lane16, hi), av);
            at = wmma16(a_hc[c], bfrag(WtT + (jn * 16) * 136 + c * 32, 136, lane16, hi), at);
        }
        int col = jn * 16 + lane16;
        float bv = bv0[col], bt = bt0[col];
#pragma unroll
        for (int p = 0; p < 8; ++p) {
            int row = row0 + wave * 16 + p + hi * 8;
            out[OFF_V0 + (size_t)row * NOPTD + col] = av[p] + bv;
            termS[(wave * 16 + p + hi * 8) * 64 + col] = sigm(at[p] + bt);
        }
    }
    __syncthreads();
    if (tid < 128) {
        int row = row0 + tid;
        int co = chosen[row] & 63;
        out[OFF_TP + row] = termS[tid * 64 + co];
    }
}

// ---------------------------------------------------------------------------
// Kernel 3: GRU layer 1. x1 = cat(attn, h1) -> K=256 for gi; gh K=128; v1 head.
// LDS: Wih1T (384x264) + Whh1T (384x136) + Wv1T (64x136) ~= 317 KB.
// ---------------------------------------------------------------------------
__global__ __launch_bounds__(256) void gru1_kernel(const _Float16* __restrict__ attn_h,
                                                   const float* __restrict__ h1,
                                                   const float* __restrict__ Wih1,
                                                   const float* __restrict__ Whh1,
                                                   const float* __restrict__ bih1,
                                                   const float* __restrict__ bhh1,
                                                   const float* __restrict__ Wv1,
                                                   const float* __restrict__ bv1,
                                                   _Float16* __restrict__ hc1_h,
                                                   float* __restrict__ out) {
    extern __shared__ _Float16 smem1[];
    _Float16* WihT = smem1;                       // 384 x 264 (K=256 + pad)
    _Float16* WhhT = WihT + 384 * 264;            // 384 x 136
    _Float16* WvT  = WhhT + 384 * 136;            // 64 x 136

    const int tid = threadIdx.x;
    const int wave = tid >> 5, lane = tid & 31;
    const int lane16 = lane & 15, hi = lane >> 4;
    const int row0 = blockIdx.x * 128;

    for (int e = tid; e < 256 * 384; e += 256) {
        int k = e / 384, n = e % 384;
        WihT[n * 264 + k] = (_Float16)Wih1[e];
    }
    for (int e = tid; e < HDIM * 384; e += 256) {
        int k = e / 384, n = e % 384;
        WhhT[n * 136 + k] = (_Float16)Whh1[e];
    }
    for (int e = tid; e < HDIM * NOPTD; e += 256) {
        int k = e >> 6, n = e & 63;
        WvT[n * 136 + k] = (_Float16)Wv1[e];
    }
    __syncthreads();

    const int arow = row0 + wave * 16 + lane16;
    v16h a_at[4], a_h1[4];
#pragma unroll
    for (int c = 0; c < 4; ++c) {
        a_at[c] = afrag(attn_h + (size_t)(row0 + wave * 16) * HDIM + c * 32, HDIM, lane16, hi);
        a_h1[c] = afrag_f32(h1 + (size_t)arow * HDIM, hi, c * 32);
    }

    for (int j = 0; j < 8; ++j) {
        v8f air = zero8(), ahr = zero8(), aiz = zero8(), ahz = zero8(), ain = zero8(), ahn = zero8();
#pragma unroll
        for (int c = 0; c < 8; ++c) {        // gi: K=256 (attn chunk then h1 chunk)
            v16h a = (c < 4) ? a_at[c] : a_h1[c - 4];
            air = wmma16(a, bfrag(WihT + (j * 16) * 264 + c * 32, 264, lane16, hi), air);
            aiz = wmma16(a, bfrag(WihT + (128 + j * 16) * 264 + c * 32, 264, lane16, hi), aiz);
            ain = wmma16(a, bfrag(WihT + (256 + j * 16) * 264 + c * 32, 264, lane16, hi), ain);
        }
#pragma unroll
        for (int c = 0; c < 4; ++c) {        // gh: K=128
            ahr = wmma16(a_h1[c], bfrag(WhhT + (j * 16) * 136 + c * 32, 136, lane16, hi), ahr);
            ahz = wmma16(a_h1[c], bfrag(WhhT + (128 + j * 16) * 136 + c * 32, 136, lane16, hi), ahz);
            ahn = wmma16(a_h1[c], bfrag(WhhT + (256 + j * 16) * 136 + c * 32, 136, lane16, hi), ahn);
        }
        int col = j * 16 + lane16;
        float bir = bih1[col],       bhr = bhh1[col];
        float biz = bih1[128 + col], bhz = bhh1[128 + col];
        float bin = bih1[256 + col], bhn = bhh1[256 + col];
#pragma unroll
        for (int p = 0; p < 8; ++p) {
            int row = row0 + wave * 16 + p + hi * 8;
            float r = sigm(air[p] + ahr[p] + bir + bhr);
            float z = sigm(aiz[p] + ahz[p] + biz + bhz);
            float n = tanh_fast(ain[p] + bin + r * (ahn[p] + bhn));
            float hv = h1[(size_t)row * HDIM + col];
            float hn = (1.0f - z) * n + z * hv;
            out[OFF_H1N + (size_t)row * HDIM + col] = hn;
            hc1_h[(size_t)row * HDIM + col] = (_Float16)((hn - 0.5f) * 0.5f + 0.5f);
        }
    }

    asm volatile("s_wait_storecnt 0x0" ::: "memory");

    v16h a_hc[4];
#pragma unroll
    for (int c = 0; c < 4; ++c)
        a_hc[c] = afrag(hc1_h + (size_t)(row0 + wave * 16) * HDIM + c * 32, HDIM, lane16, hi);

    for (int jn = 0; jn < 4; ++jn) {
        v8f av = zero8();
#pragma unroll
        for (int c = 0; c < 4; ++c)
            av = wmma16(a_hc[c], bfrag(WvT + (jn * 16) * 136 + c * 32, 136, lane16, hi), av);
        int col = jn * 16 + lane16;
        float bv = bv1[col];
#pragma unroll
        for (int p = 0; p < 8; ++p) {
            int row = row0 + wave * 16 + p + hi * 8;
            out[OFF_V1 + (size_t)row * NOPTD + col] = av[p] + bv;
        }
    }
}

extern "C" void kernel_launch(void* const* d_in, const int* in_sizes, int n_in,
                              void* d_out, int out_size, void* d_ws, size_t ws_size,
                              hipStream_t stream) {
    (void)in_sizes; (void)n_in; (void)out_size; (void)ws_size;
    const float* obs  = (const float*)d_in[0];
    const int* chosen = (const int*)d_in[1];
    const float* h0   = (const float*)d_in[2];
    const float* h1   = (const float*)d_in[3];
    const float* Wa   = (const float*)d_in[4];
    const float* ba   = (const float*)d_in[5];
    const float* Wih0 = (const float*)d_in[6];
    const float* Whh0 = (const float*)d_in[7];
    const float* bih0 = (const float*)d_in[8];
    const float* bhh0 = (const float*)d_in[9];
    const float* Wih1 = (const float*)d_in[10];
    const float* Whh1 = (const float*)d_in[11];
    const float* bih1 = (const float*)d_in[12];
    const float* bhh1 = (const float*)d_in[13];
    const float* Wv0  = (const float*)d_in[14];
    const float* bv0  = (const float*)d_in[15];
    const float* Wv1  = (const float*)d_in[16];
    const float* bv1  = (const float*)d_in[17];
    const float* Wt0  = (const float*)d_in[18];
    const float* bt0  = (const float*)d_in[19];
    float* out = (float*)d_out;

    _Float16* attn_h = (_Float16*)d_ws;
    _Float16* hc0_h  = attn_h + (size_t)BDIM * HDIM;
    _Float16* hc1_h  = hc0_h + (size_t)BDIM * HDIM;

    const int shm1 = (2 * 256 * 72 + 2 * 128 * 72) * 2;                 // 110592
    const int shm2 = (2 * 384 * 136 + 2 * 64 * 136) * 2 + 128 * 64 * 4; // 276480
    const int shm3 = (384 * 264 + 384 * 136 + 64 * 136) * 2;            // 324608
    (void)hipFuncSetAttribute(reinterpret_cast<const void*>(attn_kernel),
                              hipFuncAttributeMaxDynamicSharedMemorySize, shm1);
    (void)hipFuncSetAttribute(reinterpret_cast<const void*>(gru0_kernel),
                              hipFuncAttributeMaxDynamicSharedMemorySize, shm2);
    (void)hipFuncSetAttribute(reinterpret_cast<const void*>(gru1_kernel),
                              hipFuncAttributeMaxDynamicSharedMemorySize, shm3);

    dim3 block(256);
    attn_kernel<<<dim3(BDIM / 256), block, shm1, stream>>>(obs, Wa, ba, attn_h);
    gru0_kernel<<<dim3(BDIM / 128), block, shm2, stream>>>(attn_h, h0, Wih0, Whh0, bih0, bhh0,
                                                           Wv0, bv0, Wt0, bt0, chosen, hc0_h, out);
    gru1_kernel<<<dim3(BDIM / 128), block, shm3, stream>>>(attn_h, h1, Wih1, Whh1, bih1, bhh1,
                                                           Wv1, bv1, hc1_h, out);
}
